// OTAligner_24206435680560
// MI455X (gfx1250) — matrix operand order repaired and checked by reading, not compiled, not run on hardware
//
#include <hip/hip_runtime.h>
#include <hip/hip_bf16.h>

typedef __attribute__((ext_vector_type(16))) _Float16 v16h;
typedef __attribute__((ext_vector_type(8)))  float    v8f;
typedef __attribute__((ext_vector_type(8)))  _Float16 h8;
typedef __attribute__((ext_vector_type(4)))  _Float16 h4;

#define Bsz 8
#define Nn  2048
#define Mm  2048
#define Cc  256
#define NCHUNK 32
#define INV_EPS 20.0f          // 1/0.05
#define CLAMP_MIN 1e-8f
#define T_SCALE 16384.0f       // 2^14: lifts T (~2.4e-7) into f16 normal range
#define T_INV  (1.0f/16384.0f)

__device__ __forceinline__ float waveReduceSum(float x) {
  #pragma unroll
  for (int off = 16; off > 0; off >>= 1) x += __shfl_xor(x, off, 32);
  return x;
}

// ---------- prep: row sums of squares for feat_a/feat_b, u=v=1 ----------
__global__ __launch_bounds__(128) void prep_kernel(const float* __restrict__ fa,
                                                   const float* __restrict__ fb,
                                                   float* __restrict__ a2,
                                                   float* __restrict__ b2,
                                                   float* __restrict__ u,
                                                   float* __restrict__ v) {
  int row  = blockIdx.x * 4 + (threadIdx.x >> 5);
  int lane = threadIdx.x & 31;
  const int totalA = Bsz * Nn;
  const float* src;
  float *dst, *ones;
  int idx;
  if (row < totalA) { idx = row;           src = fa + (size_t)idx * Cc; dst = a2; ones = u; }
  else              { idx = row - totalA;  src = fb + (size_t)idx * Cc; dst = b2; ones = v; }
  float s = 0.f;
  for (int c = lane; c < Cc; c += 32) { float x = src[c]; s += x * x; }
  s = waveReduceSum(s);
  if (lane == 0) { dst[idx] = s; ones[idx] = 1.0f; }
}

// ---------- one-time f32 -> f16 feature conversion (4 elems/thread) ----------
__global__ __launch_bounds__(256) void convert_f16_kernel(const float* __restrict__ src,
                                                          _Float16* __restrict__ dst) {
  size_t base = ((size_t)blockIdx.x * 256 + threadIdx.x) * 4;
  const float4 x = *(const float4*)(src + base);
  h4 o;
  o[0] = (_Float16)x.x; o[1] = (_Float16)x.y;
  o[2] = (_Float16)x.z; o[3] = (_Float16)x.w;
  *(h4*)(dst + base) = o;
}

// ---------- one-time feat_b transpose: fbT[b][c][m] (f16), LDS-tiled ----------
__global__ __launch_bounds__(256) void transpose_b_kernel(const float* __restrict__ fb,
                                                          _Float16* __restrict__ fbT) {
  __shared__ _Float16 tile[32][33];
  int tx = threadIdx.x & 31, ty = threadIdx.x >> 5;   // 32 x 8
  int bt = blockIdx.x;
  int cT = bt % (Cc / 32); bt /= (Cc / 32);
  int mT = bt % (Mm / 32); bt /= (Mm / 32);
  int b  = bt;
  int m0 = mT * 32, c0 = cT * 32;
  const float* src = fb + (size_t)b * Mm * Cc;
  #pragma unroll
  for (int i = 0; i < 4; ++i) {
    int m = m0 + ty + i * 8;
    tile[ty + i * 8][tx] = (_Float16)src[(size_t)m * Cc + c0 + tx];
  }
  __syncthreads();
  _Float16* dst = fbT + (size_t)b * Cc * Mm;
  #pragma unroll
  for (int i = 0; i < 4; ++i) {
    int c = c0 + ty + i * 8;
    dst[(size_t)c * Mm + m0 + tx] = tile[tx][ty + i * 8];
  }
}

// ---------- cost GEMM + exp -> K (f16), one 16x16 tile per wave ----------
__global__ __launch_bounds__(128) void cost_exp_kernel(const _Float16* __restrict__ fa16,
                                                       const _Float16* __restrict__ fb16,
                                                       const float* __restrict__ a2,
                                                       const float* __restrict__ b2,
                                                       _Float16* __restrict__ Kx) {
  const int wave = threadIdx.x >> 5;
  const int lane = threadIdx.x & 31;
  const int hh   = lane >> 4;
  const int lr   = lane & 15;
  int tile = blockIdx.x * 4 + wave;
  const int tilesPerB = (Nn / 16) * (Mm / 16);
  int b   = tile / tilesPerB;
  int rem = tile % tilesPerB;
  int n0  = (rem / (Mm / 16)) * 16;
  int m0  = (rem % (Mm / 16)) * 16;

  const _Float16* arow = fa16 + ((size_t)b * Nn + n0 + lr) * Cc;
  const _Float16* brow = fb16 + ((size_t)b * Mm + m0 + lr) * Cc;

  v8f acc = {};
  for (int c0 = 0; c0 < Cc; c0 += 32) {
    // A 16x32 f16, contiguous k order chosen here: half0 -> K 0..15, half1 -> K 16..31
    h8 a0 = *(const h8*)(arow + c0 + hh * 16);
    h8 a1 = *(const h8*)(arow + c0 + hh * 16 + 8);
    v16h af = __builtin_shufflevector(a0, a1, 0,1,2,3,4,5,6,7,8,9,10,11,12,13,14,15);
    // B 32x16 f16, same contiguous k order as A: half0 -> K 0..15 of col lr, half1 -> K 16..31
    v16h bf = *(const v16h*)(brow + c0 + hh * 16);
    acc = __builtin_amdgcn_wmma_f32_16x16x32_f16(false, af, false, bf,
                                                 (short)0, acc, false, false);
  }

  const float bsq = b2[b * Mm + m0 + lr];
  _Float16* Kb = Kx + (size_t)b * Nn * Mm;
  #pragma unroll
  for (int r = 0; r < 8; ++r) {
    int n = n0 + hh * 8 + r;                       // D row = half*8 + vgpr
    float cost = a2[b * Nn + n] + bsq - 2.0f * acc[r];
    Kb[(size_t)n * Mm + m0 + lr] = (_Float16)__expf(-cost * INV_EPS);
  }
}

// ---------- Sinkhorn u-update: one wave per (b,n) row, 16B/lane loads ----------
__global__ __launch_bounds__(256) void kv_u_kernel(const _Float16* __restrict__ Kx,
                                                   const float* __restrict__ v,
                                                   float* __restrict__ u) {
  int row  = blockIdx.x * 8 + (threadIdx.x >> 5);  // [0, Bsz*Nn)
  int lane = threadIdx.x & 31;
  int b = row / Nn;
  const _Float16* Krow = Kx + (size_t)row * Mm;
  const float*    vb   = v + b * Mm;
  float s = 0.f;
  for (int m = lane * 8; m < Mm; m += 256) {
    h8 k = *(const h8*)(Krow + m);
    const float4 v0 = *(const float4*)(vb + m);
    const float4 v1 = *(const float4*)(vb + m + 4);
    s += (float)k[0] * v0.x + (float)k[1] * v0.y +
         (float)k[2] * v0.z + (float)k[3] * v0.w +
         (float)k[4] * v1.x + (float)k[5] * v1.y +
         (float)k[6] * v1.z + (float)k[7] * v1.w;
  }
  s = waveReduceSum(s);
  if (lane == 0) {
    if (!(s == s) || fabsf(s) > 1e30f) s = 1.0f;   // nan_to_num
    s = fmaxf(s, CLAMP_MIN);
    u[row] = (1.0f / (float)Nn) / s;
  }
}

// ---------- Sinkhorn v-update, stage 1: partial column sums (4 cols/thread) ----------
__global__ __launch_bounds__(256) void ku_partial_kernel(const _Float16* __restrict__ Kx,
                                                         const float* __restrict__ u,
                                                         float* __restrict__ partial) {
  int bid   = blockIdx.x;
  int mg    = bid % (Mm / 1024); bid /= (Mm / 1024);
  int chunk = bid % NCHUNK;      bid /= NCHUNK;
  int b     = bid;
  int m     = mg * 1024 + threadIdx.x * 4;
  const int rowsPer = Nn / NCHUNK;
  int n0 = chunk * rowsPer;
  const _Float16* Kb = Kx + (size_t)b * Nn * Mm;
  const float*    ub = u + b * Nn;
  float s0 = 0.f, s1 = 0.f, s2 = 0.f, s3 = 0.f;
  for (int n = n0; n < n0 + rowsPer; ++n) {
    h4 k = *(const h4*)(Kb + (size_t)n * Mm + m);
    float uu = ub[n];
    s0 += (float)k[0] * uu; s1 += (float)k[1] * uu;
    s2 += (float)k[2] * uu; s3 += (float)k[3] * uu;
  }
  *(float4*)(partial + ((size_t)b * NCHUNK + chunk) * Mm + m) =
      make_float4(s0, s1, s2, s3);
}

// ---------- Sinkhorn v-update, stage 2: combine ----------
__global__ __launch_bounds__(256) void ku_v_kernel(const float* __restrict__ partial,
                                                   float* __restrict__ v) {
  int i = blockIdx.x * 256 + threadIdx.x;          // [0, Bsz*Mm)
  int b = i / Mm, m = i % Mm;
  float s = 0.f;
  #pragma unroll
  for (int c = 0; c < NCHUNK; ++c)
    s += partial[((size_t)b * NCHUNK + c) * Mm + m];
  if (!(s == s) || fabsf(s) > 1e30f) s = 1.0f;
  s = fmaxf(s, CLAMP_MIN);
  v[i] = (1.0f / (float)Mm) / s;
}

// ---------- final: aligned = clip(u*K*v) @ feat_b via WMMA ----------
__global__ __launch_bounds__(128) void apply_kernel(const _Float16* __restrict__ Kx,
                                                    const _Float16* __restrict__ fbT,
                                                    const float* __restrict__ u,
                                                    const float* __restrict__ v,
                                                    float* __restrict__ out) {
  const int wave = threadIdx.x >> 5;
  const int lane = threadIdx.x & 31;
  const int hh   = lane >> 4;
  const int lr   = lane & 15;
  int tile = blockIdx.x * 4 + wave;
  const int tilesPerB = (Nn / 16) * (Cc / 16);
  int b   = tile / tilesPerB;
  int rem = tile % tilesPerB;
  int n0  = (rem / (Cc / 16)) * 16;
  int c0  = (rem % (Cc / 16)) * 16;

  const _Float16* Krow = Kx + ((size_t)b * Nn + n0 + lr) * Mm;   // A row per lane
  const _Float16* bcol = fbT + ((size_t)b * Cc + c0 + lr) * Mm;  // B col per lane
  const float     ul   = u[b * Nn + n0 + lr];
  const float*    vb   = v + b * Mm;

  v8f acc = {};
  for (int m0 = 0; m0 < Mm; m0 += 32) {
    int am = m0 + hh * 8;
    h8 k0 = *(const h8*)(Krow + am);
    h8 k1 = *(const h8*)(Krow + am + 16);
    const float4 va0 = *(const float4*)(vb + am);
    const float4 va1 = *(const float4*)(vb + am + 4);
    const float4 va2 = *(const float4*)(vb + am + 16);
    const float4 va3 = *(const float4*)(vb + am + 20);
    float vv0[8] = {va0.x, va0.y, va0.z, va0.w, va1.x, va1.y, va1.z, va1.w};
    float vv1[8] = {va2.x, va2.y, va2.z, va2.w, va3.x, va3.y, va3.z, va3.w};
    v16h af;
    #pragma unroll
    for (int j = 0; j < 8; ++j) {
      float t1 = fmaxf(ul * (float)k0[j] * vv0[j], CLAMP_MIN);
      float t2 = fmaxf(ul * (float)k1[j] * vv1[j], CLAMP_MIN);
      af[j]     = (_Float16)(t1 * T_SCALE);
      af[j + 8] = (_Float16)(t2 * T_SCALE);
    }
    // B fragment: half0 -> m {0..7,16..23}, half1 -> m {8..15,24..31} of column c0+lr
    h8 b0 = *(const h8*)(bcol + m0 + hh * 8);
    h8 b1 = *(const h8*)(bcol + m0 + hh * 8 + 16);
    v16h bf = __builtin_shufflevector(b0, b1, 0,1,2,3,4,5,6,7,8,9,10,11,12,13,14,15);
    acc = __builtin_amdgcn_wmma_f32_16x16x32_f16(false, af, false, bf,
                                                 (short)0, acc, false, false);
  }
  #pragma unroll
  for (int r = 0; r < 8; ++r) {
    int n = n0 + hh * 8 + r;
    out[((size_t)b * Nn + n) * Cc + c0 + lr] = acc[r] * T_INV;
  }
}

extern "C" void kernel_launch(void* const* d_in, const int* in_sizes, int n_in,
                              void* d_out, int out_size, void* d_ws, size_t ws_size,
                              hipStream_t stream) {
  const float* fa = (const float*)d_in[0];
  const float* fb = (const float*)d_in[1];
  float* out = (float*)d_out;

  char* ws = (char*)d_ws;
  size_t off = 0;
  _Float16* Kx = (_Float16*)(ws + off); off += (size_t)Bsz * Nn * Mm * sizeof(_Float16); // 64 MiB
  _Float16* fa16 = (_Float16*)(ws + off); off += (size_t)Bsz * Nn * Cc * sizeof(_Float16);
  _Float16* fb16 = (_Float16*)(ws + off); off += (size_t)Bsz * Mm * Cc * sizeof(_Float16);
  _Float16* fbT  = (_Float16*)(ws + off); off += (size_t)Bsz * Cc * Mm * sizeof(_Float16);
  float* a2 = (float*)(ws + off); off += (size_t)Bsz * Nn * sizeof(float);
  float* b2 = (float*)(ws + off); off += (size_t)Bsz * Mm * sizeof(float);
  float* u  = (float*)(ws + off); off += (size_t)Bsz * Nn * sizeof(float);
  float* v  = (float*)(ws + off); off += (size_t)Bsz * Mm * sizeof(float);
  float* partial = (float*)(ws + off); off += (size_t)Bsz * NCHUNK * Mm * sizeof(float);

  // prep: 2*Bsz*Nn rows, 4 waves/block
  prep_kernel<<<(2 * Bsz * Nn) / 4, 128, 0, stream>>>(fa, fb, a2, b2, u, v);

  // one-time f16 conversions + transpose
  convert_f16_kernel<<<((size_t)Bsz * Nn * Cc) / (256 * 4), 256, 0, stream>>>(fa, fa16);
  convert_f16_kernel<<<((size_t)Bsz * Mm * Cc) / (256 * 4), 256, 0, stream>>>(fb, fb16);
  transpose_b_kernel<<<Bsz * (Mm / 32) * (Cc / 32), 256, 0, stream>>>(fb, fbT);

  // cost GEMM + exp
  cost_exp_kernel<<<(Bsz * (Nn / 16) * (Mm / 16)) / 4, 128, 0, stream>>>(fa16, fb16, a2, b2, Kx);

  // 50 Sinkhorn iterations
  for (int it = 0; it < 50; ++it) {
    kv_u_kernel<<<(Bsz * Nn) / 8, 256, 0, stream>>>(Kx, v, u);
    ku_partial_kernel<<<Bsz * NCHUNK * (Mm / 1024), 256, 0, stream>>>(Kx, u, partial);
    ku_v_kernel<<<(Bsz * Mm) / 256, 256, 0, stream>>>(partial, v);
  }

  // final transport apply
  apply_kernel<<<(Bsz * (Nn / 16) * (Cc / 16)) / 4, 128, 0, stream>>>(Kx, fbT, u, v, out);
}
